// genes_input_12120397709336
// MI455X (gfx1250) — matrix operand order repaired and checked by reading, not compiled, not run on hardware
//
#include <hip/hip_runtime.h>
#include <stdint.h>

// Masked linear: out = x @ (W * M)^T
//   x: [4096, 8192] f32, W/M: [4096, 8192] f32, out: [4096, 4096] f32
//
// Fast path (needs 256 MB workspace):
//   1) prep_split: one-pass mask-fuse + f32 -> (bf16 hi, bf16 lo) split into d_ws
//   2) bf16-split-3 GEMM on v_wmma_f32_16x16x32_bf16, inner loop feeds LDS with
//      global_load_async_to_lds_b128 (ASYNCcnt) -- no per-iteration VALU convert.
// Fallback (small ws): fused convert-in-loop double-buffered kernel.

#define BATCH 4096
#define INS   8192
#define OUTS  4096

#define BM 128        // batch rows per workgroup
#define BN 128        // out cols per workgroup
#define BK 32         // K per iteration (one bf16 WMMA K-depth)
#define LSTRIDE 40    // u16 elements per LDS row: 32 data + 8 pad (80 B)
#define TILE (BM * LSTRIDE)

typedef __attribute__((ext_vector_type(16))) __bf16          v16bf;
typedef __attribute__((ext_vector_type(8)))  float           v8f;
typedef __attribute__((ext_vector_type(8)))  unsigned short  u16x8;
typedef __attribute__((ext_vector_type(4)))  unsigned short  u16x4;
typedef __attribute__((ext_vector_type(4)))  float           f32x4;

union FragBF { v16bf v; u16x8 h[2]; };

// Split one f32 into bf16 high part (truncation) + bf16 residual.
__device__ __forceinline__ void split4(f32x4 f, u16x4& hi, u16x4& lo) {
#pragma unroll
  for (int c = 0; c < 4; ++c) {
    unsigned u = __float_as_uint(f[c]);
    unsigned short h = (unsigned short)(u >> 16);
    float r = f[c] - __uint_as_float((unsigned)h << 16);
    hi[c] = h;
    lo[c] = (unsigned short)(__float_as_uint(r) >> 16);
  }
}

// ---------------------------------------------------------------------------
// CDNA5 async global->LDS copy: LDS[vdst_lds_byte] = MEM[saddr + vaddr], 16 B.
// VGLOBAL async encoding: VDST = LDS byte address VGPR, GVS addressing.
__device__ __forceinline__ void async_b128(unsigned lds_byte, unsigned gl_byte,
                                           const void* base) {
  asm volatile("global_load_async_to_lds_b128 %0, %1, %2"
               :
               : "v"(lds_byte), "v"(gl_byte), "s"((unsigned long long)(uintptr_t)base)
               : "memory");
}
__device__ __forceinline__ void wait_async0() {
  asm volatile("s_wait_asynccnt 0x0" ::: "memory");
}

// ---------------------------------------------------------------------------
// Kernel 1: one-pass mask fuse + bf16 hi/lo split of both operands into ws.
__global__ __launch_bounds__(256)
void prep_split(const float* __restrict__ x, const float* __restrict__ w,
                const float* __restrict__ mk,
                unsigned short* __restrict__ xhi, unsigned short* __restrict__ xlo,
                unsigned short* __restrict__ whi, unsigned short* __restrict__ wlo) {
  const size_t i = ((size_t)blockIdx.x * blockDim.x + threadIdx.x) * 4;
  u16x4 hi, lo;
  f32x4 xv = *(const f32x4*)(x + i);
  split4(xv, hi, lo);
  *(u16x4*)(xhi + i) = hi;
  *(u16x4*)(xlo + i) = lo;
  f32x4 wv = *(const f32x4*)(w + i);
  f32x4 mv = *(const f32x4*)(mk + i);
  split4(wv * mv, hi, lo);                      // fused ontology mask
  *(u16x4*)(whi + i) = hi;
  *(u16x4*)(wlo + i) = lo;
}

// ---------------------------------------------------------------------------
// Kernel 2 (fast path): pure-bf16 GEMM, async global->LDS double-buffered.
__global__ __launch_bounds__(256)
void gemm_bf16x3_async(const unsigned short* __restrict__ xhi,
                       const unsigned short* __restrict__ xlo,
                       const unsigned short* __restrict__ whi,
                       const unsigned short* __restrict__ wlo,
                       float* __restrict__ out) {
  __shared__ unsigned short lAhi[2 * TILE];
  __shared__ unsigned short lAlo[2 * TILE];
  __shared__ unsigned short lBhi[2 * TILE];
  __shared__ unsigned short lBlo[2 * TILE];

  const int t    = threadIdx.x;
  const int lane = t & 31;
  const int wave = t >> 5;
  const int wm   = wave >> 2;   // 0..1  -> 64-row slab of C
  const int wn   = wave & 3;    // 0..3  -> 32-col slab of C

  const int m0 = blockIdx.y * BM;
  const int n0 = blockIdx.x * BN;

  v8f acc[4][2] = {};

  // Fragment addressing (ISA 7.12.2 16-bit layouts), as before.
  const int frow  = lane & 15;
  const int fhalf = lane >> 4;
  const int aoff  = 8 * fhalf;
  const int boff  = 8 * fhalf;

  // Async loader mapping: per tile kind 512 x 16B chunks; 2 chunks/thread.
  // chunk = t + 256*j : row = chunk>>2 (0..127), col = (chunk&3)*8 bf16.
  auto issue_tile = [&](int buf, int kt) {
#pragma unroll
    for (int j = 0; j < 2; ++j) {
      const int ch = t + 256 * j;
      const int r  = ch >> 2;
      const int c8 = (ch & 3) * 8;
      const unsigned ga = (unsigned)((((size_t)(m0 + r) * INS) + kt + c8) * 2);
      const unsigned gw = (unsigned)((((size_t)(n0 + r) * INS) + kt + c8) * 2);
      const int li = buf * TILE + r * LSTRIDE + c8;
      async_b128((unsigned)(uintptr_t)(void*)&lAhi[li], ga, xhi);
      async_b128((unsigned)(uintptr_t)(void*)&lAlo[li], ga, xlo);
      async_b128((unsigned)(uintptr_t)(void*)&lBhi[li], gw, whi);
      async_b128((unsigned)(uintptr_t)(void*)&lBlo[li], gw, wlo);
    }
  };

  // Prologue: tile 0 -> buffer 0.
  issue_tile(0, 0);
  wait_async0();
  __syncthreads();

  int cur = 0;
  for (int kt = 0; kt < INS; kt += BK) {
    const bool has_next = (kt + BK) < INS;
    if (has_next) issue_tile(cur ^ 1, kt + BK);   // DMA next tile during WMMA

    const unsigned short* Ahi = &lAhi[cur * TILE];
    const unsigned short* Alo = &lAlo[cur * TILE];
    const unsigned short* Bhi = &lBhi[cur * TILE];
    const unsigned short* Blo = &lBlo[cur * TILE];

    FragBF bhi[2], blo[2];
#pragma unroll
    for (int in = 0; in < 2; ++in) {
      const int r = wn * 32 + in * 16 + frow;
      const unsigned short* p = &Bhi[r * LSTRIDE + boff];
      bhi[in].h[0] = *(const u16x8*)p;
      bhi[in].h[1] = *(const u16x8*)(p + 16);
      const unsigned short* q = &Blo[r * LSTRIDE + boff];
      blo[in].h[0] = *(const u16x8*)q;
      blo[in].h[1] = *(const u16x8*)(q + 16);
    }
#pragma unroll
    for (int im = 0; im < 4; ++im) {
      const int r = wm * 64 + im * 16 + frow;
      FragBF ahi, alo;
      const unsigned short* p = &Ahi[r * LSTRIDE + aoff];
      ahi.h[0] = *(const u16x8*)p;
      ahi.h[1] = *(const u16x8*)(p + 16);
      const unsigned short* q = &Alo[r * LSTRIDE + aoff];
      alo.h[0] = *(const u16x8*)q;
      alo.h[1] = *(const u16x8*)(q + 16);
#pragma unroll
      for (int in = 0; in < 2; ++in) {
        acc[im][in] = __builtin_amdgcn_wmma_f32_16x16x32_bf16(
            false, ahi.v, false, bhi[in].v, (short)0, acc[im][in], false, false);
        acc[im][in] = __builtin_amdgcn_wmma_f32_16x16x32_bf16(
            false, ahi.v, false, blo[in].v, (short)0, acc[im][in], false, false);
        acc[im][in] = __builtin_amdgcn_wmma_f32_16x16x32_bf16(
            false, alo.v, false, bhi[in].v, (short)0, acc[im][in], false, false);
      }
    }

    if (has_next) {
      wait_async0();        // own async writes done
      __syncthreads();      // everyone's async writes visible
    }
    cur ^= 1;
  }

  const int cn  = lane & 15;
  const int cm0 = (lane >> 4) * 8;
#pragma unroll
  for (int im = 0; im < 4; ++im) {
    const int gm = m0 + wm * 64 + im * 16 + cm0;
#pragma unroll
    for (int in = 0; in < 2; ++in) {
      const int gn = n0 + wn * 32 + in * 16 + cn;
      float* op = out + (size_t)gm * OUTS + gn;
#pragma unroll
      for (int v = 0; v < 8; ++v)
        op[(size_t)v * OUTS] = acc[im][in][v];
    }
  }
}

// ---------------------------------------------------------------------------
// Fallback: fused convert-in-loop double-buffered kernel.
struct Stage {
  f32x4 xa[4];
  f32x4 wv[4];
  f32x4 mv[4];
};

__device__ __forceinline__ Stage load_stage(const float* __restrict__ x,
                                            const float* __restrict__ w,
                                            const float* __restrict__ mk,
                                            size_t xbase, size_t wbase, int kt) {
  Stage s;
#pragma unroll
  for (int pass = 0; pass < 4; ++pass) {
    const size_t go = (size_t)pass * 32 * INS + kt;
    s.xa[pass] = *(const f32x4*)(x  + xbase + go);
    s.wv[pass] = *(const f32x4*)(w  + wbase + go);
    s.mv[pass] = *(const f32x4*)(mk + wbase + go);
  }
  return s;
}

__global__ __launch_bounds__(256)
void masked_linear_bf16x3_fused(const float* __restrict__ x,
                                const float* __restrict__ w,
                                const float* __restrict__ mk,
                                float* __restrict__ out) {
  __shared__ unsigned short lAhi[2 * TILE];
  __shared__ unsigned short lAlo[2 * TILE];
  __shared__ unsigned short lBhi[2 * TILE];
  __shared__ unsigned short lBlo[2 * TILE];

  const int t    = threadIdx.x;
  const int lane = t & 31;
  const int wave = t >> 5;
  const int wm   = wave >> 2;
  const int wn   = wave & 3;

  const int m0 = blockIdx.y * BM;
  const int n0 = blockIdx.x * BN;

  const int lr = t >> 3;
  const int lc = (t & 7) * 4;

  const size_t xbase = (size_t)(m0 + lr) * INS + lc;
  const size_t wbase = (size_t)(n0 + lr) * INS + lc;

  v8f acc[4][2] = {};

  const int frow  = lane & 15;
  const int fhalf = lane >> 4;
  const int aoff  = 8 * fhalf;
  const int boff  = 8 * fhalf;

  {
    Stage s0 = load_stage(x, w, mk, xbase, wbase, 0);
#pragma unroll
    for (int pass = 0; pass < 4; ++pass) {
      const int idx = (lr + pass * 32) * LSTRIDE + lc;
      u16x4 hi, lo;
      split4(s0.xa[pass], hi, lo);
      *(u16x4*)&lAhi[idx] = hi;
      *(u16x4*)&lAlo[idx] = lo;
      split4(s0.wv[pass] * s0.mv[pass], hi, lo);
      *(u16x4*)&lBhi[idx] = hi;
      *(u16x4*)&lBlo[idx] = lo;
    }
  }
  __syncthreads();

  int cur = 0;
  for (int kt = 0; kt < INS; kt += BK) {
    const bool has_next = (kt + BK) < INS;

    Stage nxt;
    if (has_next) nxt = load_stage(x, w, mk, xbase, wbase, kt + BK);

    const unsigned short* Ahi = &lAhi[cur * TILE];
    const unsigned short* Alo = &lAlo[cur * TILE];
    const unsigned short* Bhi = &lBhi[cur * TILE];
    const unsigned short* Blo = &lBlo[cur * TILE];

    FragBF bhi[2], blo[2];
#pragma unroll
    for (int in = 0; in < 2; ++in) {
      const int r = wn * 32 + in * 16 + frow;
      const unsigned short* p = &Bhi[r * LSTRIDE + boff];
      bhi[in].h[0] = *(const u16x8*)p;
      bhi[in].h[1] = *(const u16x8*)(p + 16);
      const unsigned short* q = &Blo[r * LSTRIDE + boff];
      blo[in].h[0] = *(const u16x8*)q;
      blo[in].h[1] = *(const u16x8*)(q + 16);
    }
#pragma unroll
    for (int im = 0; im < 4; ++im) {
      const int r = wm * 64 + im * 16 + frow;
      FragBF ahi, alo;
      const unsigned short* p = &Ahi[r * LSTRIDE + aoff];
      ahi.h[0] = *(const u16x8*)p;
      ahi.h[1] = *(const u16x8*)(p + 16);
      const unsigned short* q = &Alo[r * LSTRIDE + aoff];
      alo.h[0] = *(const u16x8*)q;
      alo.h[1] = *(const u16x8*)(q + 16);
#pragma unroll
      for (int in = 0; in < 2; ++in) {
        acc[im][in] = __builtin_amdgcn_wmma_f32_16x16x32_bf16(
            false, ahi.v, false, bhi[in].v, (short)0, acc[im][in], false, false);
        acc[im][in] = __builtin_amdgcn_wmma_f32_16x16x32_bf16(
            false, ahi.v, false, blo[in].v, (short)0, acc[im][in], false, false);
        acc[im][in] = __builtin_amdgcn_wmma_f32_16x16x32_bf16(
            false, alo.v, false, bhi[in].v, (short)0, acc[im][in], false, false);
      }
    }

    if (has_next) {
      const int nb = (cur ^ 1) * TILE;
#pragma unroll
      for (int pass = 0; pass < 4; ++pass) {
        const int idx = nb + (lr + pass * 32) * LSTRIDE + lc;
        u16x4 hi, lo;
        split4(nxt.xa[pass], hi, lo);
        *(u16x4*)&lAhi[idx] = hi;
        *(u16x4*)&lAlo[idx] = lo;
        split4(nxt.wv[pass] * nxt.mv[pass], hi, lo);
        *(u16x4*)&lBhi[idx] = hi;
        *(u16x4*)&lBlo[idx] = lo;
      }
      __syncthreads();
    }
    cur ^= 1;
  }

  const int cn  = lane & 15;
  const int cm0 = (lane >> 4) * 8;
#pragma unroll
  for (int im = 0; im < 4; ++im) {
    const int gm = m0 + wm * 64 + im * 16 + cm0;
#pragma unroll
    for (int in = 0; in < 2; ++in) {
      const int gn = n0 + wn * 32 + in * 16 + cn;
      float* op = out + (size_t)gm * OUTS + gn;
#pragma unroll
      for (int v = 0; v < 8; ++v)
        op[(size_t)v * OUTS] = acc[im][in][v];
    }
  }
}

// ---------------------------------------------------------------------------
extern "C" void kernel_launch(void* const* d_in, const int* in_sizes, int n_in,
                              void* d_out, int out_size, void* d_ws, size_t ws_size,
                              hipStream_t stream) {
  const float* x  = (const float*)d_in[0];
  const float* w  = (const float*)d_in[1];
  const float* mk = (const float*)d_in[2];
  float* out = (float*)d_out;
  (void)in_sizes; (void)n_in; (void)out_size;

  const size_t elems = (size_t)BATCH * INS;            // == OUTS * INS
  const size_t need  = 4 * elems * sizeof(unsigned short);  // 256 MB
  dim3 grid(OUTS / BN, BATCH / BM);                    // (32, 32)

  if (ws_size >= need) {
    unsigned short* xhi = (unsigned short*)d_ws;
    unsigned short* xlo = xhi + elems;
    unsigned short* whi = xlo + elems;
    unsigned short* wlo = whi + elems;
    prep_split<<<(unsigned)(elems / 4 / 256), 256, 0, stream>>>(x, w, mk, xhi, xlo, whi, wlo);
    gemm_bf16x3_async<<<grid, 256, 0, stream>>>(xhi, xlo, whi, wlo, out);
  } else {
    masked_linear_bf16x3_fused<<<grid, 256, 0, stream>>>(x, w, mk, out);
  }
}
